// DiscriminatorDecider_11759620457093
// MI455X (gfx1250) — matrix operand order repaired and checked by reading, not compiled, not run on hardware
//
#include <hip/hip_runtime.h>
#include <hip/hip_bf16.h>
#include <math.h>

typedef __attribute__((ext_vector_type(16))) _Float16 v16h;
typedef __attribute__((ext_vector_type(8)))  _Float16 v8h;
typedef __attribute__((ext_vector_type(8)))  float    v8f;

#define LRELU_SLOPE 0.2f
#define BN_EPS 1e-5f
#define SN_EPS 1e-12f

// ---------------------------------------------------------------------------
// f32 -> f16 conversion
// ---------------------------------------------------------------------------
__global__ void cvt_f16_kernel(const float* __restrict__ src,
                               _Float16* __restrict__ dst, int n) {
  int i = blockIdx.x * 256 + threadIdx.x;
  if (i < n) dst[i] = (_Float16)src[i];
}

// T (f32, [256][4096]) -> Tt (f16, [4096][256]) so B fragments load contiguously
__global__ void transpose_T_kernel(const float* __restrict__ T,
                                   _Float16* __restrict__ Tt) {
  int idx = blockIdx.x * 256 + threadIdx.x;  // total 4096*256
  int j = idx >> 8;       // column of the product, 0..4095
  int k = idx & 255;      // K index, 0..255
  Tt[idx] = (_Float16)T[k * 4096 + j];
}

// ---------------------------------------------------------------------------
// Spectral norm: one block per layer.  sigma = ||W v||^2 / (||W v|| + eps),
// v = W^T u / (||W^T u|| + eps).  Writes f16 weights [co][Kpad], zero padded.
// ---------------------------------------------------------------------------
__global__ void sn_kernel(const float* __restrict__ w, const float* __restrict__ u,
                          _Float16* __restrict__ w16, int co, int K, int Kpad) {
  __shared__ float vsh[8192];
  __shared__ float wv[512];
  __shared__ float red[256];
  int t = threadIdx.x;  // 256 threads

  // v = W^T u
  for (int j = t; j < K; j += 256) {
    float a = 0.f;
    for (int i = 0; i < co; ++i) a += w[i * K + j] * u[i];
    vsh[j] = a;
  }
  __syncthreads();

  // ||v||
  float p = 0.f;
  for (int j = t; j < K; j += 256) p += vsh[j] * vsh[j];
  red[t] = p;
  __syncthreads();
  for (int s = 128; s > 0; s >>= 1) {
    if (t < s) red[t] += red[t + s];
    __syncthreads();
  }
  float inv_nv = 1.0f / (sqrtf(red[0]) + SN_EPS);
  __syncthreads();
  for (int j = t; j < K; j += 256) vsh[j] *= inv_nv;
  __syncthreads();

  // Wv
  for (int i = t; i < co; i += 256) {
    float a = 0.f;
    for (int j = 0; j < K; ++j) a += w[i * K + j] * vsh[j];
    wv[i] = a;
  }
  __syncthreads();

  // sigma
  p = 0.f;
  for (int i = t; i < co; i += 256) p += wv[i] * wv[i];
  red[t] = p;
  __syncthreads();
  for (int s = 128; s > 0; s >>= 1) {
    if (t < s) red[t] += red[t + s];
    __syncthreads();
  }
  float n2 = red[0];                      // ||Wv||^2
  float sigma = n2 / (sqrtf(n2) + SN_EPS);
  float invs = 1.0f / sigma;

  int tot = co * Kpad;
  for (int idx = t; idx < tot; idx += 256) {
    int i = idx / Kpad;
    int k = idx - i * Kpad;
    w16[idx] = (_Float16)((k < K) ? w[i * K + k] * invs : 0.0f);
  }
}

// ---------------------------------------------------------------------------
// Implicit-GEMM conv (k=4, s=2, p=1) via v_wmma_f32_16x16x32_f16.
// Block = 128 threads = 4 waves.  Macro-tile: 64 pixels x 64 couts.
// Each wave owns 16 couts (one A fragment per K step) and all 4 pixel groups
// (4 B fragments -> 4 independent WMMA accumulation chains).
// Weight K-chunks are staged into LDS with GLOBAL_LOAD_ASYNC_TO_LDS_B128
// (ASYNCcnt) and double-buffered so the async DMA for chunk c+1 overlaps the
// WMMA compute on chunk c.  im2col tile is gathered with normal loads
// (needs boundary zero fill) and double-buffered the same way.
// ---------------------------------------------------------------------------
template <bool FUSE_LRELU_F16>
__global__ void conv_wmma(const _Float16* __restrict__ x,
                          const _Float16* __restrict__ w16,
                          float* __restrict__ yraw,
                          _Float16* __restrict__ yact,
                          int Cin, int H, int W, int Cout,
                          int OH, int OW, int Kpad) {
  __shared__ __align__(16) _Float16 tile[2][64 * 40];   // 64 px x 32 K (+pad)
  __shared__ __align__(16) _Float16 wtile[2][64 * 32];  // 64 couts x 32 K
  __shared__ int pn[64], poy[64], pox[64];

  const int t    = threadIdx.x;
  const int lane = t & 31;
  const int wave = t >> 5;
  const int l16  = lane & 15;
  const bool hi  = lane >= 16;

  const int pbase   = blockIdx.x * 64;
  const int coutBlk = blockIdx.y * 64;

  if (t < 64) {
    int p = pbase + t;
    int ohw = OH * OW;
    int n = p / ohw;
    int s = p - n * ohw;
    pn[t]  = n;
    poy[t] = s / OW;
    pox[t] = s - (s / OW) * OW;
  }
  __syncthreads();

  // Stage one 32-K chunk into LDS buffer `buf`:
  //  - weights via per-lane async-to-LDS b128 copies (always in-bounds)
  //  - im2col gather via regular loads + ds stores (boundary zero fill)
  auto stage = [&](int buf, int kb) {
    for (int tid = t; tid < 256; tid += 128) {   // 256 x 16B = 4KB weight chunk
      int row = tid >> 2;       // cout row within block tile
      int seg = tid & 3;        // 16-byte segment within the 64B row chunk
      unsigned lds = (unsigned)(uintptr_t)&wtile[buf][row * 32 + seg * 8];
      unsigned long long g = (unsigned long long)(
          w16 + (size_t)(coutBlk + row) * Kpad + kb + seg * 8);
      asm volatile("global_load_async_to_lds_b128 %0, %1, off"
                   :: "v"(lds), "v"(g) : "memory");
    }
    for (int idx = t; idx < 64 * 32; idx += 128) {
      int pix = idx >> 5;
      int kk  = idx & 31;
      int k   = kb + kk;
      _Float16 val = (_Float16)0.f;
      int ci = k >> 4;
      if (ci < Cin) {
        int r  = k & 15;
        int ky = r >> 2;
        int kx = r & 3;
        int iy = poy[pix] * 2 - 1 + ky;
        int ix = pox[pix] * 2 - 1 + kx;
        if ((unsigned)iy < (unsigned)H && (unsigned)ix < (unsigned)W)
          val = x[((pn[pix] * Cin + ci) * H + iy) * W + ix];
      }
      tile[buf][pix * 40 + kk] = val;
    }
  };

  v8f acc0 = {}, acc1 = {}, acc2 = {}, acc3 = {};

  stage(0, 0);
  asm volatile("s_wait_asynccnt 0x0" ::: "memory");
  __syncthreads();

  const int nchunks = Kpad >> 5;
  for (int c = 0; c < nchunks; ++c) {
    int cur = c & 1;
    if (c + 1 < nchunks) stage(cur ^ 1, (c + 1) << 5);

    // A fragment: 16-bit A 16x32 layout.  lane<16: K {0..7,16..23}; hi: +8.
    const _Float16* wr = &wtile[cur][(wave * 16 + l16) * 32 + (hi ? 8 : 0)];
    v8h a0 = *(const v8h*)wr;
    v8h a1 = *(const v8h*)(wr + 16);
    v16h A = __builtin_shufflevector(a0, a1, 0, 1, 2, 3, 4, 5, 6, 7,
                                     8, 9, 10, 11, 12, 13, 14, 15);

    // B fragments: 16-bit B 32x16 layout.  col = lane&15; lane<16: K {0..7,16..23}; hi: +8.
#define CONV_B_FRAG(G)                                                      \
    const _Float16* br##G = &tile[cur][((G) * 16 + l16) * 40 + (hi ? 8 : 0)]; \
    v8h b0##G = *(const v8h*)br##G;                                         \
    v8h b1##G = *(const v8h*)(br##G + 16);                                  \
    v16h B##G = __builtin_shufflevector(b0##G, b1##G, 0, 1, 2, 3, 4, 5, 6, 7, \
                                        8, 9, 10, 11, 12, 13, 14, 15);
    CONV_B_FRAG(0); CONV_B_FRAG(1); CONV_B_FRAG(2); CONV_B_FRAG(3);
#undef CONV_B_FRAG

    acc0 = __builtin_amdgcn_wmma_f32_16x16x32_f16(false, A, false, B0,
                                                  (short)0, acc0, false, false);
    acc1 = __builtin_amdgcn_wmma_f32_16x16x32_f16(false, A, false, B1,
                                                  (short)0, acc1, false, false);
    acc2 = __builtin_amdgcn_wmma_f32_16x16x32_f16(false, A, false, B2,
                                                  (short)0, acc2, false, false);
    acc3 = __builtin_amdgcn_wmma_f32_16x16x32_f16(false, A, false, B3,
                                                  (short)0, acc3, false, false);

    asm volatile("s_wait_asynccnt 0x0" ::: "memory");
    __syncthreads();
  }

  // Epilogue: C/D layout — n = lane&15 (pixel col), m = r + (hi?8:0) (cout row)
  const int OHW = OH * OW;
#pragma unroll
  for (int g = 0; g < 4; ++g) {
    const v8f& acc = (g == 0) ? acc0 : (g == 1) ? acc1 : (g == 2) ? acc2 : acc3;
    int px = g * 16 + l16;
    int n  = pn[px];
    int sp = poy[px] * OW + pox[px];
#pragma unroll
    for (int r = 0; r < 8; ++r) {
      int m  = r + (hi ? 8 : 0);
      int cc = coutBlk + wave * 16 + m;
      int idx = (n * Cout + cc) * OHW + sp;
      if constexpr (FUSE_LRELU_F16) {
        float v = acc[r];
        v = (v >= 0.f) ? v : LRELU_SLOPE * v;
        yact[idx] = (_Float16)v;
      } else {
        yraw[idx] = acc[r];
      }
    }
  }
}

// ---------------------------------------------------------------------------
// BatchNorm (training-mode batch stats): per-channel reduce then apply+lrelu
// ---------------------------------------------------------------------------
__global__ void bn_stats(const float* __restrict__ raw,
                         const float* __restrict__ g, const float* __restrict__ b,
                         float* __restrict__ scale, float* __restrict__ shift,
                         int C, int Nb, int HW) {
  __shared__ float rs[256], rq[256];
  int c = blockIdx.x;
  int t = threadIdx.x;
  int total = Nb * HW;
  float s = 0.f, q = 0.f;
  for (int i = t; i < total; i += 256) {
    int n  = i / HW;
    int sp = i - n * HW;
    float v = raw[(n * C + c) * HW + sp];
    s += v;
    q += v * v;
  }
  rs[t] = s;
  rq[t] = q;
  __syncthreads();
  for (int st = 128; st > 0; st >>= 1) {
    if (t < st) { rs[t] += rs[t + st]; rq[t] += rq[t + st]; }
    __syncthreads();
  }
  if (t == 0) {
    float mean = rs[0] / (float)total;
    float var  = rq[0] / (float)total - mean * mean;
    float rstd = rsqrtf(var + BN_EPS);
    float sc = rstd * g[c];
    scale[c] = sc;
    shift[c] = b[c] - mean * sc;
  }
}

__global__ void bn_apply(const float* __restrict__ raw,
                         const float* __restrict__ scale,
                         const float* __restrict__ shift,
                         _Float16* __restrict__ act, int C, int HW) {
  int idx = blockIdx.x * 256 + threadIdx.x;
  int c = (idx / HW) % C;
  float v = raw[idx] * scale[c] + shift[c];
  v = (v >= 0.f) ? v : LRELU_SLOPE * v;
  act[idx] = (_Float16)v;
}

// ---------------------------------------------------------------------------
// Ms = flat[64,256] @ T[256,4096] via WMMA.  One wave per 16x16 tile.
// grid = (4096/16, 64/16), block = 32.
// ---------------------------------------------------------------------------
__global__ void gemm_ms(const _Float16* __restrict__ flat16,
                        const _Float16* __restrict__ Tt,
                        float* __restrict__ Ms) {
  int lane = threadIdx.x;
  int l16  = lane & 15;
  bool hi  = lane >= 16;
  int cb = blockIdx.x * 16;  // column tile
  int rb = blockIdx.y * 16;  // batch-row tile

  v8f acc = {};
  for (int kb = 0; kb < 256; kb += 32) {
    const _Float16* ar = flat16 + (rb + l16) * 256 + kb + (hi ? 8 : 0);
    v8h a0 = *(const v8h*)ar;
    v8h a1 = *(const v8h*)(ar + 16);
    const _Float16* br = Tt + (cb + l16) * 256 + kb + (hi ? 8 : 0);
    v8h b0 = *(const v8h*)br;
    v8h b1 = *(const v8h*)(br + 16);
    v16h A = __builtin_shufflevector(a0, a1, 0, 1, 2, 3, 4, 5, 6, 7,
                                     8, 9, 10, 11, 12, 13, 14, 15);
    v16h B = __builtin_shufflevector(b0, b1, 0, 1, 2, 3, 4, 5, 6, 7,
                                     8, 9, 10, 11, 12, 13, 14, 15);
    acc = __builtin_amdgcn_wmma_f32_16x16x32_f16(false, A, false, B,
                                                 (short)0, acc, false, false);
  }
#pragma unroll
  for (int r = 0; r < 8; ++r) {
    int m = r + (hi ? 8 : 0);
    Ms[(rb + m) * 4096 + cb + l16] = acc[r];
  }
}

// ---------------------------------------------------------------------------
// Minibatch discrimination: out[i, 256+b] = sum_j exp(-sum_c |Ms[i,b,c]-Ms[j,b,c]|)
// grid = (64, 64) = (i, b); block = 64 threads (one per j)
// ---------------------------------------------------------------------------
__global__ void mbd_kernel(const float* __restrict__ Ms, float* __restrict__ out) {
  __shared__ float mi[64];
  __shared__ float red[64];
  int i = blockIdx.x, b = blockIdx.y;
  int j = threadIdx.x;
  mi[j] = Ms[i * 4096 + b * 64 + j];
  __syncthreads();
  const float* mj = Ms + j * 4096 + b * 64;
  float d = 0.f;
#pragma unroll 8
  for (int c = 0; c < 64; ++c) d += fabsf(mi[c] - mj[c]);
  red[j] = __expf(-d);
  __syncthreads();
  for (int s = 32; s > 0; s >>= 1) {
    if (j < s) red[j] += red[j + s];
    __syncthreads();
  }
  if (j == 0) out[i * 320 + 256 + b] = red[0];
}

__global__ void copy_flat(const float* __restrict__ flat, float* __restrict__ out) {
  int idx = blockIdx.x * 256 + threadIdx.x;  // 64*256
  int i = idx >> 8;
  int a = idx & 255;
  out[i * 320 + a] = flat[idx];
}

// ---------------------------------------------------------------------------
extern "C" void kernel_launch(void* const* d_in, const int* in_sizes, int n_in,
                              void* d_out, int out_size, void* d_ws, size_t ws_size,
                              hipStream_t stream) {
  (void)in_sizes; (void)n_in; (void)out_size; (void)ws_size;

  const float* image = (const float*)d_in[0];
  const float* Wt[6], *Uv[6];
  for (int l = 0; l < 6; ++l) {
    Wt[l] = (const float*)d_in[1 + 2 * l];
    Uv[l] = (const float*)d_in[2 + 2 * l];
  }
  const float* G[4], *Bb[4];
  for (int l = 0; l < 4; ++l) {
    G[l]  = (const float*)d_in[13 + 2 * l];
    Bb[l] = (const float*)d_in[14 + 2 * l];
  }
  const float* T = (const float*)d_in[21];
  float* out = (float*)d_out;
  char* ws = (char*)d_ws;

  const int Nb = 64;
  const int Cin[6]  = {3, 64, 128, 256, 256, 512};
  const int Cout[6] = {64, 128, 256, 256, 512, 64};
  const int Hin[6]  = {128, 64, 32, 16, 8, 4};
  const int Kk[6]   = {48, 1024, 2048, 4096, 4096, 8192};
  const int Kpad[6] = {64, 1024, 2048, 4096, 4096, 8192};

  size_t off = 0;
  auto alloc = [&](size_t bytes) {
    size_t o = off;
    off += (bytes + 255) & ~(size_t)255;
    return o;
  };

  _Float16* img16 = (_Float16*)(ws + alloc((size_t)Nb * 3 * 128 * 128 * 2));
  _Float16* w16p[6];
  for (int l = 0; l < 6; ++l)
    w16p[l] = (_Float16*)(ws + alloc((size_t)Cout[l] * Kpad[l] * 2));
  _Float16* actp[6];  // f16 activations after layer l (l=0..4 used as inputs)
  float* rawp[6];     // raw f32 conv outputs (l=1..5 used)
  for (int l = 0; l < 6; ++l) {
    int OHW = (Hin[l] / 2) * (Hin[l] / 2);
    actp[l] = (_Float16*)(ws + alloc((size_t)Nb * Cout[l] * OHW * 2));
    rawp[l] = (float*)(ws + alloc((size_t)Nb * Cout[l] * OHW * 4));
  }
  float* scalep[4];
  float* shiftp[4];
  for (int l = 0; l < 4; ++l) {
    scalep[l] = (float*)(ws + alloc((size_t)Cout[l + 1] * 4));
    shiftp[l] = (float*)(ws + alloc((size_t)Cout[l + 1] * 4));
  }
  _Float16* flat16 = (_Float16*)(ws + alloc((size_t)Nb * 256 * 2));
  _Float16* Tt     = (_Float16*)(ws + alloc((size_t)4096 * 256 * 2));
  float* Ms        = (float*)(ws + alloc((size_t)Nb * 4096 * 4));

  // image -> f16
  {
    int n = Nb * 3 * 128 * 128;
    cvt_f16_kernel<<<(n + 255) / 256, 256, 0, stream>>>(image, img16, n);
  }
  // spectral-norm + f16 weight repack
  for (int l = 0; l < 6; ++l)
    sn_kernel<<<1, 256, 0, stream>>>(Wt[l], Uv[l], w16p[l], Cout[l], Kk[l], Kpad[l]);

  // conv layers: macro-tile 64 pixels x 64 couts per 128-thread block
  for (int l = 0; l < 6; ++l) {
    int H = Hin[l], OH = H / 2, OW = OH;
    int P = Nb * OH * OW;
    dim3 grid(P / 64, Cout[l] / 64);
    const _Float16* xin = (l == 0) ? img16 : actp[l - 1];
    if (l == 0) {
      conv_wmma<true><<<grid, 128, 0, stream>>>(
          xin, w16p[0], nullptr, actp[0], Cin[0], H, H, Cout[0], OH, OW, Kpad[0]);
    } else {
      conv_wmma<false><<<grid, 128, 0, stream>>>(
          xin, w16p[l], rawp[l], nullptr, Cin[l], H, H, Cout[l], OH, OW, Kpad[l]);
      if (l <= 4) {
        int HW = OH * OW;
        bn_stats<<<Cout[l], 256, 0, stream>>>(rawp[l], G[l - 1], Bb[l - 1],
                                              scalep[l - 1], shiftp[l - 1],
                                              Cout[l], Nb, HW);
        int total = Nb * Cout[l] * HW;
        bn_apply<<<total / 256, 256, 0, stream>>>(rawp[l], scalep[l - 1],
                                                  shiftp[l - 1], actp[l],
                                                  Cout[l], HW);
      }
    }
  }

  // flat = conv6 raw output ([64,64,2,2] NCHW == [64,256] row-major)
  const float* flat = rawp[5];
  transpose_T_kernel<<<4096, 256, 0, stream>>>(T, Tt);
  cvt_f16_kernel<<<64, 256, 0, stream>>>(flat, flat16, Nb * 256);
  gemm_ms<<<dim3(4096 / 16, Nb / 16), 32, 0, stream>>>(flat16, Tt, Ms);
  copy_flat<<<64, 256, 0, stream>>>(flat, out);
  mbd_kernel<<<dim3(Nb, Nb), 64, 0, stream>>>(Ms, out);
}
